// ButterFlyNet2D_10617159155724
// MI455X (gfx1250) — matrix operand order repaired and checked by reading, not compiled, not run on hardware
//
#include <hip/hip_runtime.h>
#include <hip/hip_bf16.h>
#include <stdint.h>

// ---------------------------------------------------------------------------
// ButterFlyNet2D forward for MI455X (gfx1250).
//  B=32, C=1, H=W=64, L=6, T=4  ->  N=64, KO=256, first kernel 2x2, lk=1.
//  State layout between layers: [gy][gx][y][x][n=64][b=32] fp32.
//  Every state has 64*64 * 64 * 32 = 8,388,608 floats (33.5 MB); ping-pong in ws.
// ---------------------------------------------------------------------------

#define NB    32
#define NCH   64
#define KOUT  256
#define STATE_ELEMS (4096u * NCH * NB)   // 8,388,608 floats per state buffer
#define A_STRIDE 36                      // LDS row stride (dwords): 32 data + 4 pad (TDM pad)

typedef __attribute__((ext_vector_type(16))) __bf16 v16bf;
typedef __attribute__((ext_vector_type(8)))  float  v8f;
typedef __attribute__((ext_vector_type(4)))  unsigned int u32x4;
typedef __attribute__((ext_vector_type(8)))  int          i32x8;
typedef __attribute__((ext_vector_type(4)))  int          i32x4;

// ---------------------------------------------------------------------------
// TDM: DMA a 2D tile (256 rows x 32 fp32) of the weight matrix into LDS.
// Row stride in memory = 256 elements; LDS gets +4 dwords pad per 32-dword row
// (pad_interval code 4 = 32 dwords, pad_amount code 3 = 4 dwords) -> A_STRIDE=36.
// ---------------------------------------------------------------------------
__device__ __forceinline__ void tdm_load_panel(const float* gsrc, unsigned int lds_off)
{
    uint64_t ga = (uint64_t)(uintptr_t)gsrc;
    u32x4 g0;
    g0.x = 1u;                                            // count=1 (user descriptor)
    g0.y = lds_off;                                       // lds_addr [63:32]
    g0.z = (unsigned int)(ga & 0xffffffffu);              // global_addr [95:64]
    g0.w = (unsigned int)((ga >> 32) & 0x01ffffffu)       // global_addr [120:96]
         | (2u << 30);                                    // type=2 ("image")

    i32x8 g1;
    g1[0] = (int)((2u << 16)        // data_size = 4 bytes
                | (1u << 20)        // pad_enable
                | (4u << 22)        // pad_interval: 32 dwords
                | (3u << 25));      // pad_amount:   4 dwords
    g1[1] = (int)(256u << 16);      // tensor_dim0 = 256 (low16 @ [63:48])
    g1[2] = (int)(256u << 16);      // tensor_dim0 hi=0, tensor_dim1 = 256 (low16 @ [95:80])
    g1[3] = (int)(32u << 16);       // tensor_dim1 hi=0, tile_dim0 = 32
    g1[4] = (int)256;               // tile_dim1 = 256, tile_dim2 = 0
    g1[5] = (int)256;               // tensor_dim0_stride = 256 elements
    g1[6] = 0;
    g1[7] = 0;

    i32x4 gz = {0, 0, 0, 0};
#if defined(__clang_major__) && (__clang_major__ >= 23)
    i32x8 gz8 = {0, 0, 0, 0, 0, 0, 0, 0};
    __builtin_amdgcn_tensor_load_to_lds(g0, g1, gz, gz, gz8, 0);
#else
    __builtin_amdgcn_tensor_load_to_lds(g0, g1, gz, gz, 0);
#endif
}

__device__ __forceinline__ v16bf cvt16(float4 a, float4 b, float4 c, float4 d)
{
    v16bf r;
    r[0]  = (__bf16)a.x; r[1]  = (__bf16)a.y; r[2]  = (__bf16)a.z; r[3]  = (__bf16)a.w;
    r[4]  = (__bf16)b.x; r[5]  = (__bf16)b.y; r[6]  = (__bf16)b.z; r[7]  = (__bf16)b.w;
    r[8]  = (__bf16)c.x; r[9]  = (__bf16)c.y; r[10] = (__bf16)c.z; r[11] = (__bf16)c.w;
    r[12] = (__bf16)d.x; r[13] = (__bf16)d.y; r[14] = (__bf16)d.z; r[15] = (__bf16)d.w;
    return r;
}

// ---------------------------------------------------------------------------
// First layer: input has only component 0 nonzero -> 4-tap conv per out chan.
// One WG per (Y,X) of the 32x32 grid. k = ((ly*2+lx)*64 + n).
// ---------------------------------------------------------------------------
__global__ __launch_bounds__(256) void bfly_first_layer(
    const float* __restrict__ in,    // [32][64][64]
    const float* __restrict__ W0,    // [256][4][2][2]
    const float* __restrict__ b0,    // [256]
    float* __restrict__ st)          // [2][2][32][32][64][32]
{
    __shared__ float pix[NB][4];
    const int X = blockIdx.x & 31, Y = blockIdx.x >> 5;
    const int t = threadIdx.x;
    if (t < 128) {
        int b = t >> 2, i = t & 3, h = i >> 1, w = i & 1;
        pix[b][i] = in[b * 4096 + (2 * Y + h) * 64 + (2 * X + w)];
    }
    __syncthreads();
    const int k  = t;
    const float w0 = W0[k * 16 + 0], w1 = W0[k * 16 + 1];
    const float w2 = W0[k * 16 + 2], w3 = W0[k * 16 + 3];
    const float bias = b0[k];
    const int cell = k >> 6;          // ly*2+lx
    const int n    = k & 63;
    float* dst = st + ((size_t)cell * 1024 + (size_t)Y * 32 + X) * (NCH * NB) + n * 32;
#pragma unroll 4
    for (int b = 0; b < NB; ++b) {
        float acc = bias + pix[b][0] * w0 + pix[b][1] * w1 + pix[b][2] * w2 + pix[b][3] * w3;
        dst[b] = fmaxf(acc, 0.f);
    }
}

// ---------------------------------------------------------------------------
// Recursion layer: per (u,v,Yo,Xo) a GEMM  D[256x32] = W[256x256] * B[256x32].
// K index kk = n*4 + p*2 + q (contiguous in memory along weight rows).
// A (weights fp32) streamed via TDM into double-buffered LDS panels, converted
// to bf16 in registers; B staged once as bf16 (transposed [col][kk]).
// 8 waves, each owns M-tiles {2w,2w+1} x N-tiles {0,1}: 4 WMMAs / k-step.
// ---------------------------------------------------------------------------
__global__ __launch_bounds__(256) void bfly_recursion(
    const float* __restrict__ stin,  // [G][G][s][s][64][32]
    const float* __restrict__ Wl,    // [G][G][256][64][2][2]
    const float* __restrict__ bl,    // [G][G][256]
    float* __restrict__ stout,       // [2G][2G][s2][s2][64][32]
    int G, int s)
{
    const int s2 = s >> 1;
    __shared__ __align__(128) float  Apan[2][KOUT * A_STRIDE]; // 2 x 36 KB
    __shared__ __align__(128) __bf16 Bmat[NB][KOUT];           // 16 KB, [col][kk]

    const int u = blockIdx.x / G, v = blockIdx.x % G;
    const int Yo = blockIdx.y / s2, Xo = blockIdx.y % s2;
    const int t = threadIdx.x;
    const float* Wblk = Wl + (size_t)(u * G + v) * (KOUT * NCH * 4);

    // ---- stage B (im2col of activations) into LDS as bf16 ----
    {
        const size_t blkbase = (size_t)(u * G + v) * ((size_t)s * s * (NCH * NB));
#pragma unroll
        for (int e = t; e < KOUT * NB; e += 256) {
            int kk = e >> 5, col = e & 31;
            int n = kk >> 2, p = (kk >> 1) & 1, q = kk & 1;
            float x = stin[blkbase + (size_t)((2 * Yo + p) * s + (2 * Xo + q)) * (NCH * NB)
                           + n * 32 + col];
            Bmat[col][kk] = (__bf16)x;
        }
    }

    const int wave = t >> 5, lane = t & 31;
    const int mrow = lane & 15, ksel = lane >> 4;
    v8f acc[2][2];
#pragma unroll
    for (int i = 0; i < 2; ++i)
#pragma unroll
        for (int j = 0; j < 2; ++j)
#pragma unroll
            for (int r = 0; r < 8; ++r) acc[i][j][r] = 0.f;

    // prime the TDM pipeline (one wave issues; DMA lands in shared LDS)
    if (wave == 0)
        tdm_load_panel(Wblk, (unsigned int)(uintptr_t)&Apan[0][0]);

    for (int kp = 0; kp < 8; ++kp) {
        if (wave == 0) {
            if (kp < 7) {
                tdm_load_panel(Wblk + (kp + 1) * 32,
                               (unsigned int)(uintptr_t)&Apan[(kp + 1) & 1][0]);
                __builtin_amdgcn_s_wait_tensorcnt(1);   // panel kp complete
            } else {
                __builtin_amdgcn_s_wait_tensorcnt(0);
            }
        }
        __syncthreads();
        const float* Ap = &Apan[kp & 1][0];

        // A fragments (16x32 bf16): lane L -> M=L&15, K = ksel*16 .. +15
        v16bf afrag[2];
#pragma unroll
        for (int mt = 0; mt < 2; ++mt) {
            int m = (2 * wave + mt) * 16 + mrow;
            const float* row = Ap + m * A_STRIDE;
            float4 c0 = *(const float4*)(row + ksel * 16);
            float4 c1 = *(const float4*)(row + ksel * 16 + 4);
            float4 c2 = *(const float4*)(row + ksel * 16 + 8);
            float4 c3 = *(const float4*)(row + ksel * 16 + 12);
            afrag[mt] = cvt16(c0, c1, c2, c3);
        }
        // B fragments (32x16 bf16): lane L -> N=L&15, K = (L>>4)*16 .. +15
        v16bf bfrag[2];
#pragma unroll
        for (int nt = 0; nt < 2; ++nt) {
            int col = nt * 16 + (lane & 15);
            int kkbase = kp * 32 + (lane >> 4) * 16;
            bfrag[nt] = *(const v16bf*)&Bmat[col][kkbase];
        }
#pragma unroll
        for (int mt = 0; mt < 2; ++mt)
#pragma unroll
            for (int nt = 0; nt < 2; ++nt)
                acc[mt][nt] = __builtin_amdgcn_wmma_f32_16x16x32_bf16(
                    false, afrag[mt], false, bfrag[nt],
                    (short)0, acc[mt][nt], false, false);
        __syncthreads();
    }

    // ---- epilogue: bias + ReLU + butterfly scatter ----
    const float* brow = bl + (size_t)(u * G + v) * KOUT;
    const int G2 = 2 * G;
#pragma unroll
    for (int mt = 0; mt < 2; ++mt) {
#pragma unroll
        for (int nt = 0; nt < 2; ++nt) {
#pragma unroll
            for (int r = 0; r < 8; ++r) {
                int mloc = r + 8 * (lane >> 4);
                int k = (2 * wave + mt) * 16 + mloc;
                int b = nt * 16 + (lane & 15);
                int ly = k >> 7, lx = (k >> 6) & 1, n = k & 63;
                float val = fmaxf(acc[mt][nt][r] + brow[k], 0.f);
                size_t idx = ((size_t)(((2 * u + ly) * G2 + (2 * v + lx)) * s2 + Yo) * s2 + Xo)
                                 * (NCH * NB) + n * 32 + b;
                stout[idx] = val;
            }
        }
    }
}

// ---------------------------------------------------------------------------
// Final 1x1 layer + joint(): out[b][0][comp][u][v] = relu-dots, c0-c2 / c1-c3.
// ---------------------------------------------------------------------------
__global__ __launch_bounds__(128) void bfly_final(
    const float* __restrict__ st,   // [64][64][64][32]
    const float* __restrict__ Wf,   // [64][64][4][64]
    const float* __restrict__ bf,   // [64][64][4]
    float* __restrict__ out)        // [32][1][2][64][64]
{
    __shared__ float sblk[NCH * NB];
    __shared__ float ybuf[NB][4];
    const int u = blockIdx.x >> 6, v = blockIdx.x & 63;
    const int t = threadIdx.x;
    const float* src = st + (size_t)(u * 64 + v) * (NCH * NB);
    for (int e = t; e < NCH * NB; e += 128) sblk[e] = src[e];
    __syncthreads();
    const int b = t >> 2, k = t & 3;
    const float* wrow = Wf + (size_t)(u * 64 + v) * 256 + k * 64;
    float acc = bf[(u * 64 + v) * 4 + k];
#pragma unroll
    for (int n = 0; n < NCH; ++n) acc += sblk[n * 32 + b] * wrow[n];
    ybuf[b][k] = fmaxf(acc, 0.f);
    __syncthreads();
    if (t < 64) {
        int bb = t >> 1, comp = t & 1;
        float val = ybuf[bb][comp] - ybuf[bb][comp + 2];
        out[((size_t)(bb * 2 + comp) * 64 + u) * 64 + v] = val;
    }
}

// ---------------------------------------------------------------------------
extern "C" void kernel_launch(void* const* d_in, const int* in_sizes, int n_in,
                              void* d_out, int out_size, void* d_ws, size_t ws_size,
                              hipStream_t stream)
{
    (void)in_sizes; (void)n_in; (void)out_size; (void)ws_size;
    const float* input = (const float*)d_in[0];
    const float* W0 = (const float*)d_in[1];
    const float* b0 = (const float*)d_in[2];
    const float* Wl[5]  = { (const float*)d_in[3], (const float*)d_in[5],
                            (const float*)d_in[7], (const float*)d_in[9],
                            (const float*)d_in[11] };
    const float* bll[5] = { (const float*)d_in[4], (const float*)d_in[6],
                            (const float*)d_in[8], (const float*)d_in[10],
                            (const float*)d_in[12] };
    const float* Wf = (const float*)d_in[13];
    const float* bf = (const float*)d_in[14];

    float* buf0 = (float*)d_ws;
    float* buf1 = buf0 + STATE_ELEMS;

    bfly_first_layer<<<dim3(1024), dim3(256), 0, stream>>>(input, W0, b0, buf0);

    float* cur = buf0;
    float* nxt = buf1;
    int G = 2, s = 32;
    for (int l = 0; l < 5; ++l) {
        dim3 grid(G * G, (s / 2) * (s / 2));
        bfly_recursion<<<grid, dim3(256), 0, stream>>>(cur, Wl[l], bll[l], nxt, G, s);
        float* tmp = cur; cur = nxt; nxt = tmp;
        G *= 2; s /= 2;
    }

    bfly_final<<<dim3(4096), dim3(128), 0, stream>>>(cur, Wf, bf, (float*)d_out);
}
